// SelfAttentionLayer_39041252721096
// MI455X (gfx1250) — matrix operand order repaired and checked by reading, not compiled, not run on hardware
//
#include <hip/hip_runtime.h>
#include <hip/hip_bf16.h>

typedef __bf16 v16bf __attribute__((ext_vector_type(16)));
typedef float  v8f   __attribute__((ext_vector_type(8)));

#define C_DIM 256
#define N_TOK 1024
#define N_BATCH 32
#define TOK_TOTAL (N_BATCH * N_TOK)   // 32768
#define ROW_TILES (TOK_TOTAL / 16)    // 2048
#define COL_TILES (C_DIM / 16)        // 16

union BF16x16 {
    v16bf v;
    __bf16 h[16];
    unsigned short s[16];
    uint4 q[2];
};

// native fptrunc -> hardware bf16 convert (RNE), instead of manual bit math
__device__ __forceinline__ unsigned short f2bf(float f) {
    __bf16 b = (__bf16)f;
    return __builtin_bit_cast(unsigned short, b);
}

__device__ __forceinline__ v8f wmma_bf16(const BF16x16& a, const BF16x16& b, v8f c) {
    return __builtin_amdgcn_wmma_f32_16x16x32_bf16(
        /*neg_a=*/false, a.v, /*neg_b=*/false, b.v,
        /*c_mod=*/(short)0, c, /*reuse_a=*/false, /*reuse_b=*/false);
}

// ---------------------------------------------------------------------------
// Kernel 0: transpose-convert weights fp32 [k][n] -> bf16 [n][k]; fold 1/16 into Wq
// grid = 4*256 blocks of 256 threads
// ---------------------------------------------------------------------------
__global__ __launch_bounds__(256) void prep_weights(
    const float* __restrict__ Wq, const float* __restrict__ Wk,
    const float* __restrict__ Wv, const float* __restrict__ Wo,
    unsigned short* __restrict__ WqT, unsigned short* __restrict__ WkT,
    unsigned short* __restrict__ WvT, unsigned short* __restrict__ WoT)
{
    int m = blockIdx.x >> 8;     // which matrix
    int n = blockIdx.x & 255;    // output row = original column
    int k = threadIdx.x;         // 256 threads = K dim
    const float* W = (m == 0) ? Wq : (m == 1) ? Wk : (m == 2) ? Wv : Wo;
    unsigned short* WT = (m == 0) ? WqT : (m == 1) ? WkT : (m == 2) ? WvT : WoT;
    float sc = (m == 0) ? 0.0625f : 1.0f;   // 1/sqrt(256) folded into Wq
    WT[n * C_DIM + k] = f2bf(W[k * C_DIM + n] * sc);
}

// ---------------------------------------------------------------------------
// Kernel 1: QKV projection. One wave per 16x16 output tile per projection.
// q,k row-major bf16 [tok][C]; v transposed bf16 [b][c][n].
// grid = 3*ROW_TILES*COL_TILES/8 = 12288 blocks of 256 threads (8 waves)
// ---------------------------------------------------------------------------
__global__ __launch_bounds__(256) void qkv_proj(
    const float* __restrict__ x,
    const unsigned short* __restrict__ WqT, const unsigned short* __restrict__ WkT,
    const unsigned short* __restrict__ WvT,
    const float* __restrict__ bq, const float* __restrict__ bk, const float* __restrict__ bv,
    unsigned short* __restrict__ qOut, unsigned short* __restrict__ kOut,
    unsigned short* __restrict__ vT)
{
    int wave = (blockIdx.x * blockDim.x + threadIdx.x) >> 5;
    int lane = threadIdx.x & 31;
    int proj = wave / (ROW_TILES * COL_TILES);
    int rem  = wave % (ROW_TILES * COL_TILES);
    int rt = rem >> 4;
    int ct = rem & 15;

    const unsigned short* WT = (proj == 0) ? WqT : (proj == 1) ? WkT : WvT;
    const float* bias = (proj == 0) ? bq : (proj == 1) ? bk : bv;
    float bscale = (proj == 0) ? 0.0625f : 1.0f;

    int l15 = lane & 15;
    int hi  = lane >> 4;
    int kbaseA = hi * 8;     // A operand K base per lane half
    int khalfB = hi * 8;     // B operand K base per lane half
    int row = rt * 16 + l15; // token (A row)
    int col = ct * 16 + l15; // output channel (B col / D col)

    const float* xrow = x + (size_t)row * C_DIM;
    const unsigned short* wrow = WT + (size_t)col * C_DIM;

    v8f acc = {};
    for (int ks = 0; ks < 8; ++ks) {
        int k0 = ks * 32;
        BF16x16 a, b;
        const float* p = xrow + k0 + kbaseA;
        float4 f0 = *(const float4*)(p);
        float4 f1 = *(const float4*)(p + 4);
        float4 f2 = *(const float4*)(p + 16);
        float4 f3 = *(const float4*)(p + 20);
        a.h[0] = (__bf16)f0.x; a.h[1] = (__bf16)f0.y; a.h[2] = (__bf16)f0.z; a.h[3] = (__bf16)f0.w;
        a.h[4] = (__bf16)f1.x; a.h[5] = (__bf16)f1.y; a.h[6] = (__bf16)f1.z; a.h[7] = (__bf16)f1.w;
        a.h[8] = (__bf16)f2.x; a.h[9] = (__bf16)f2.y; a.h[10]= (__bf16)f2.z; a.h[11]= (__bf16)f2.w;
        a.h[12]= (__bf16)f3.x; a.h[13]= (__bf16)f3.y; a.h[14]= (__bf16)f3.z; a.h[15]= (__bf16)f3.w;
        b.q[0] = *(const uint4*)(wrow + k0 + khalfB);
        b.q[1] = *(const uint4*)(wrow + k0 + khalfB + 16);
        acc = wmma_bf16(a, b, acc);
    }

    float bval = bias[col] * bscale;
    if (proj < 2) {
        unsigned short* out = (proj == 0) ? qOut : kOut;
        int rbase = rt * 16 + hi * 8;
        for (int r = 0; r < 8; ++r)
            out[(size_t)(rbase + r) * C_DIM + col] = f2bf(acc[r] + bval);
    } else {
        // v transposed: vT[b][c][n], 8 contiguous tokens per lane -> one b128 store
        int g0 = rt * 16;
        int b_ = g0 / N_TOK;
        int n0 = (g0 % N_TOK) + hi * 8;
        BF16x16 st;
        for (int r = 0; r < 8; ++r) st.h[r] = (__bf16)(acc[r] + bval);
        *(uint4*)(vT + ((size_t)b_ * C_DIM + col) * N_TOK + n0) = st.q[0];
    }
}

// ---------------------------------------------------------------------------
// Kernel 2: attention. One workgroup per (batch, 16-row tile).
// LDS: scores f32 16x1024 (64KB) + P bf16 16x1024 (32KB) + reductions.
// grid = 2048 blocks of 256 threads, dynamic LDS.
// ---------------------------------------------------------------------------
#define ATTN_SMEM_BYTES (16*1024*4 + 16*1024*2 + 16*16*4 + 16*4 + 16*4)

__global__ __launch_bounds__(256) void attention(
    const unsigned short* __restrict__ q,    // [tok][C], scale pre-folded
    const unsigned short* __restrict__ k,    // [tok][C]
    const unsigned short* __restrict__ vT,   // [b][c][n]
    unsigned short* __restrict__ attOut)     // [tok][C]
{
    extern __shared__ char smem[];
    float*          sS   = (float*)smem;                          // 64KB scores
    unsigned short* sP   = (unsigned short*)(smem + 65536);       // 32KB exp(bf16)
    float*          sRed = (float*)(smem + 98304);                // 16x16 partials
    float*          sMax = (float*)(smem + 99328);
    float*          sInv = (float*)(smem + 99392);

    int b_ = blockIdx.x >> 6;    // batch
    int nt = blockIdx.x & 63;    // row tile
    int n0 = nt * 16;

    int tid  = threadIdx.x;
    int wid  = tid >> 5;
    int lane = tid & 31;
    int l15 = lane & 15;
    int hi  = lane >> 4;
    int kbaseA = hi * 8;
    int khalfB = hi * 8;

    const unsigned short* qrow = q + ((size_t)b_ * N_TOK + n0 + l15) * C_DIM;

    // -------- Phase 1: scores = q @ k^T (scale already folded into q) --------
    for (int mt = wid; mt < 64; mt += 8) {
        const unsigned short* krow = k + ((size_t)b_ * N_TOK + mt * 16 + l15) * C_DIM;
        v8f acc = {};
        for (int ks = 0; ks < 8; ++ks) {
            int k0 = ks * 32;
            BF16x16 a, b;
            a.q[0] = *(const uint4*)(qrow + k0 + kbaseA);
            a.q[1] = *(const uint4*)(qrow + k0 + kbaseA + 16);
            b.q[0] = *(const uint4*)(krow + k0 + khalfB);
            b.q[1] = *(const uint4*)(krow + k0 + khalfB + 16);
            acc = wmma_bf16(a, b, acc);
        }
        for (int r = 0; r < 8; ++r)
            sS[(r + hi * 8) * 1024 + mt * 16 + l15] = acc[r];
    }
    __syncthreads();

    // -------- Phase 2: softmax over 1024 cols per row (16 threads/row) --------
    {
        int row = tid & 15;
        int sub = tid >> 4;
        float mx = -3.0e38f;
        for (int j = 0; j < 64; ++j)
            mx = fmaxf(mx, sS[row * 1024 + sub + j * 16]);
        sRed[row * 16 + sub] = mx;
        __syncthreads();
        if (sub == 0) {
            float m2 = sRed[row * 16];
            for (int i = 1; i < 16; ++i) m2 = fmaxf(m2, sRed[row * 16 + i]);
            sMax[row] = m2;
        }
        __syncthreads();
        mx = sMax[row];
        float psum = 0.0f;
        for (int j = 0; j < 64; ++j) {
            int m = sub + j * 16;
            float e = __expf(sS[row * 1024 + m] - mx);
            psum += e;
            sP[row * 1024 + m] = f2bf(e);
        }
        sRed[row * 16 + sub] = psum;
        __syncthreads();
        if (sub == 0) {
            float s2 = 0.0f;
            for (int i = 0; i < 16; ++i) s2 += sRed[row * 16 + i];
            sInv[row] = 1.0f / s2;
        }
        __syncthreads();
    }

    // -------- Phase 3: attended = P @ V (K = 1024), 2 col-tiles per wave --------
    const unsigned short* prow = sP + (size_t)l15 * 1024;
    for (int ci = 0; ci < 2; ++ci) {
        int ct = wid + ci * 8;
        int colc = ct * 16 + l15;
        const unsigned short* vrow = vT + ((size_t)b_ * C_DIM + colc) * N_TOK;
        v8f acc = {};
        for (int ks = 0; ks < 32; ++ks) {
            int m0 = ks * 32;
            BF16x16 a, b;
            a.q[0] = *(const uint4*)(prow + m0 + kbaseA);        // ds_load_b128
            a.q[1] = *(const uint4*)(prow + m0 + kbaseA + 16);
            b.q[0] = *(const uint4*)(vrow + m0 + khalfB);
            b.q[1] = *(const uint4*)(vrow + m0 + khalfB + 16);
            acc = wmma_bf16(a, b, acc);
        }
        for (int r = 0; r < 8; ++r) {
            float val = acc[r] * sInv[r + hi * 8];
            attOut[((size_t)b_ * N_TOK + n0 + r + hi * 8) * C_DIM + colc] = f2bf(val);
        }
    }
}

// ---------------------------------------------------------------------------
// Kernel 3: out = attended @ Wo + bo  (fp32 output)
// grid = ROW_TILES*COL_TILES/8 = 4096 blocks of 256 threads
// ---------------------------------------------------------------------------
__global__ __launch_bounds__(256) void out_proj(
    const unsigned short* __restrict__ att, const unsigned short* __restrict__ WoT,
    const float* __restrict__ bo, float* __restrict__ out)
{
    int wave = (blockIdx.x * blockDim.x + threadIdx.x) >> 5;
    int lane = threadIdx.x & 31;
    int rt = wave >> 4;
    int ct = wave & 15;
    int l15 = lane & 15;
    int hi  = lane >> 4;
    int kbaseA = hi * 8;
    int khalfB = hi * 8;
    int row = rt * 16 + l15;
    int col = ct * 16 + l15;

    const unsigned short* arow = att + (size_t)row * C_DIM;
    const unsigned short* wrow = WoT + (size_t)col * C_DIM;

    v8f acc = {};
    for (int ks = 0; ks < 8; ++ks) {
        int k0 = ks * 32;
        BF16x16 a, b;
        a.q[0] = *(const uint4*)(arow + k0 + kbaseA);
        a.q[1] = *(const uint4*)(arow + k0 + kbaseA + 16);
        b.q[0] = *(const uint4*)(wrow + k0 + khalfB);
        b.q[1] = *(const uint4*)(wrow + k0 + khalfB + 16);
        acc = wmma_bf16(a, b, acc);
    }
    float bval = bo[col];
    int rbase = rt * 16 + hi * 8;
    for (int r = 0; r < 8; ++r)
        out[(size_t)(rbase + r) * C_DIM + col] = acc[r] + bval;
}

// ---------------------------------------------------------------------------
extern "C" void kernel_launch(void* const* d_in, const int* in_sizes, int n_in,
                              void* d_out, int out_size, void* d_ws, size_t ws_size,
                              hipStream_t stream) {
    const float* x  = (const float*)d_in[0];
    const float* Wq = (const float*)d_in[1];
    const float* bq = (const float*)d_in[2];
    const float* Wk = (const float*)d_in[3];
    const float* bk = (const float*)d_in[4];
    const float* Wv = (const float*)d_in[5];
    const float* bv = (const float*)d_in[6];
    const float* Wo = (const float*)d_in[7];
    const float* bo = (const float*)d_in[8];
    float* out = (float*)d_out;

    // workspace layout (bf16 = unsigned short)
    const size_t TOKC = (size_t)TOK_TOTAL * C_DIM;   // 8,388,608 elems
    unsigned short* ws  = (unsigned short*)d_ws;
    unsigned short* qB  = ws;
    unsigned short* kB  = ws + TOKC;
    unsigned short* vT  = ws + 2 * TOKC;
    unsigned short* att = ws + 3 * TOKC;
    unsigned short* WqT = ws + 4 * TOKC;
    unsigned short* WkT = WqT + C_DIM * C_DIM;
    unsigned short* WvT = WkT + C_DIM * C_DIM;
    unsigned short* WoT = WvT + C_DIM * C_DIM;

    prep_weights<<<4 * C_DIM, 256, 0, stream>>>(Wq, Wk, Wv, Wo, WqT, WkT, WvT, WoT);

    qkv_proj<<<3 * ROW_TILES * COL_TILES / 8, 256, 0, stream>>>(
        x, WqT, WkT, WvT, bq, bk, bv, qB, kB, vT);

    attention<<<N_BATCH * (N_TOK / 16), 256, ATTN_SMEM_BYTES, stream>>>(
        qB, kB, vT, att);

    out_proj<<<ROW_TILES * COL_TILES / 8, 256, 0, stream>>>(att, WoT, bo, out);
}
